// GINClassifier_35527969472945
// MI455X (gfx1250) — matrix operand order repaired and checked by reading, not compiled, not run on hardware
//
#include <hip/hip_runtime.h>

#define N_NODES 50000
#define N_EDGES 800000
#define HDIM    128
#define GDIM    128
#define CDIM    10

typedef __attribute__((ext_vector_type(16))) __bf16 v16bf;
typedef __attribute__((ext_vector_type(8)))  float  v8f;

union BFrag { v16bf v; unsigned int u[8]; };

__device__ __forceinline__ unsigned short f2bf(float f) {
    unsigned int u = __float_as_uint(f);
    u += 0x7FFFu + ((u >> 16) & 1u);   // round-to-nearest-even
    return (unsigned short)(u >> 16);
}
__device__ __forceinline__ unsigned int pack2bf(float lo, float hi) {
    return (unsigned int)f2bf(lo) | ((unsigned int)f2bf(hi) << 16);
}

// ---------------------------------------------------------------------------
// Pack fp32 weight [128][128] (row-major, K-major) into bf16 pair-packed
// B-fragment layout: out[kp*128 + n] = {bf16(w[2kp][n]), bf16(w[2kp+1][n])}
// ---------------------------------------------------------------------------
__global__ void pack_weight_kernel(const float* __restrict__ w,
                                   unsigned int* __restrict__ out) {
    int tid = blockIdx.x * blockDim.x + threadIdx.x;   // 64*128 = 8192
    int kp = tid >> 7;
    int n  = tid & 127;
    float lo = w[(2 * kp)     * HDIM + n];
    float hi = w[(2 * kp + 1) * HDIM + n];
    out[kp * HDIM + n] = pack2bf(lo, hi);
}

// agg[i] = (1 + eps[layer]) * h[i]   (vectorized float4)
__global__ void init_agg_kernel(const float* __restrict__ h,
                                const float* __restrict__ eps, int layer,
                                float* __restrict__ agg) {
    int i = blockIdx.x * blockDim.x + threadIdx.x;     // N*H/4 elements
    float s = 1.0f + eps[layer];
    float4 v = ((const float4*)h)[i];
    float4 r = { s * v.x, s * v.y, s * v.z, s * v.w };
    ((float4*)agg)[i] = r;
}

// One wave per edge: agg[dst[e]] += h[src[e]]  (128 floats, 4 per lane)
__global__ void scatter_kernel(const float* __restrict__ h,
                               const int* __restrict__ src,
                               const int* __restrict__ dst,
                               float* __restrict__ agg) {
    int tid = blockIdx.x * blockDim.x + threadIdx.x;
    int e = tid >> 5;
    if (e >= N_EDGES) return;
    int c = (tid & 31) * 4;
    int s = src[e], d = dst[e];
    float4 v = *(const float4*)(h + (size_t)s * HDIM + c);
    float* p = agg + (size_t)d * HDIM + c;
    unsafeAtomicAdd(p + 0, v.x);
    unsafeAtomicAdd(p + 1, v.y);
    unsafeAtomicAdd(p + 2, v.z);
    unsafeAtomicAdd(p + 3, v.w);
}

// ---------------------------------------------------------------------------
// Fused GIN MLP: hout = relu( relu(x@W1 + b1) @ W2 + b2 )
// Block = 256 threads (8 waves) covers a 32-node tile (two 16-row A tiles).
// Wave w owns output columns 16w..16w+15; each B fragment feeds 2 WMMAs.
// Last block clamps loads / masks stores (50000 = 1562*32 + 16).
// ---------------------------------------------------------------------------
__global__ void __launch_bounds__(256) gin_mlp_kernel(
    const float* __restrict__ x,          // [N,128] pre-combined (1+eps)h+agg
    const unsigned int* __restrict__ w1pk, const float* __restrict__ b1,
    const unsigned int* __restrict__ w2pk, const float* __restrict__ b2,
    float* __restrict__ hout) {

    __shared__ unsigned int  lds_x[32 * 64];                 // bf16 pairs [32][128]
    __shared__ __align__(4) unsigned short lds_y[32 * 128];  // bf16       [32][128]

    const int t  = threadIdx.x;
    const int m0 = blockIdx.x * 32;

    // Stage x tile -> bf16 LDS (row-major). 256 threads x 16 floats = 32x128.
    {
        int row = t >> 3;                      // 0..31
        int cb  = (t & 7) * 16;                // column base
        int rowg = m0 + row;
        if (rowg >= N_NODES) rowg = N_NODES - 1;   // clamp (stores masked later)
        const float* p = x + (size_t)rowg * HDIM + cb;
        int pi = row * 64 + (cb >> 1);
        #pragma unroll
        for (int q = 0; q < 4; ++q) {
            float4 a = *(const float4*)(p + 4 * q);
            lds_x[pi + 2 * q + 0] = pack2bf(a.x, a.y);
            lds_x[pi + 2 * q + 1] = pack2bf(a.z, a.w);
        }
    }
    __syncthreads();

    const int wave = t >> 5;
    const int lane = t & 31;
    const int half = lane >> 4;       // 0: lanes 0-15, 1: lanes 16-31
    const int mrow = lane & 15;       // A-matrix row within tile
    const int n    = wave * 16 + (lane & 15);   // output column

    // ---- GEMM1: y = x @ W1 (two 16-row tiles share each B fragment) -------
    v8f acc0 = {}, acc1 = {};
    #pragma unroll
    for (int kb = 0; kb < HDIM; kb += 32) {
        BFrag a0, a1, b;
        #pragma unroll
        for (int j = 0; j < 8; ++j) {
            // A 16x32 bf16 layout: VGPR j -> K = kb + (j/4)*16 + half*8 + (j%4)*2
            int k = kb + ((j >> 2) << 4) + (half << 3) + ((j & 3) << 1);
            a0.u[j] = lds_x[mrow * 64 + (k >> 1)];
            a1.u[j] = lds_x[(mrow + 16) * 64 + (k >> 1)];
            // B 32x16 bf16 layout: VGPR j -> K rows (kb + (j/4)*16 + half*8 + (j%4)*2, +1)
            int kp = (kb >> 1) + ((j >> 2) << 3) + (half << 2) + (j & 3);
            b.u[j] = w1pk[kp * HDIM + n];
        }
        acc0 = __builtin_amdgcn_wmma_f32_16x16x32_bf16(
                   false, a0.v, false, b.v, (short)0, acc0, false, false);
        acc1 = __builtin_amdgcn_wmma_f32_16x16x32_bf16(
                   false, a1.v, false, b.v, (short)0, acc1, false, false);
    }
    float bias1 = b1[n];
    #pragma unroll
    for (int r = 0; r < 8; ++r) {
        float v0 = acc0[r] + bias1;                 // C/D: M = r + 8*half
        float v1 = acc1[r] + bias1;
        v0 = v0 > 0.0f ? v0 : 0.0f;
        v1 = v1 > 0.0f ? v1 : 0.0f;
        lds_y[(r + (half << 3)) * HDIM + n]        = f2bf(v0);
        lds_y[(r + (half << 3) + 16) * HDIM + n]   = f2bf(v1);
    }
    __syncthreads();

    // ---- GEMM2: h = relu(y) @ W2 ------------------------------------------
    const unsigned int* ly = (const unsigned int*)lds_y;
    v8f acc2 = {}, acc3 = {};
    #pragma unroll
    for (int kb = 0; kb < HDIM; kb += 32) {
        BFrag a0, a1, b;
        #pragma unroll
        for (int j = 0; j < 8; ++j) {
            int k = kb + ((j >> 2) << 4) + (half << 3) + ((j & 3) << 1);
            a0.u[j] = ly[mrow * 64 + (k >> 1)];
            a1.u[j] = ly[(mrow + 16) * 64 + (k >> 1)];
            int kp = (kb >> 1) + ((j >> 2) << 3) + (half << 2) + (j & 3);
            b.u[j] = w2pk[kp * HDIM + n];
        }
        acc2 = __builtin_amdgcn_wmma_f32_16x16x32_bf16(
                   false, a0.v, false, b.v, (short)0, acc2, false, false);
        acc3 = __builtin_amdgcn_wmma_f32_16x16x32_bf16(
                   false, a1.v, false, b.v, (short)0, acc3, false, false);
    }
    float bias2 = b2[n];
    #pragma unroll
    for (int r = 0; r < 8; ++r) {
        int row0 = m0 + r + (half << 3);
        int row1 = row0 + 16;
        float v0 = acc2[r] + bias2;
        float v1 = acc3[r] + bias2;
        v0 = v0 > 0.0f ? v0 : 0.0f;
        v1 = v1 > 0.0f ? v1 : 0.0f;
        if (row0 < N_NODES) hout[(size_t)row0 * HDIM + n] = v0;
        if (row1 < N_NODES) hout[(size_t)row1 * HDIM + n] = v1;
    }
}

__global__ void zero_kernel(float* __restrict__ p, int n) {
    int i = blockIdx.x * blockDim.x + threadIdx.x;
    if (i < n) p[i] = 0.0f;
}

// One wave per node: hg[graph_ids[n]] += h[n]
__global__ void pool_kernel(const float* __restrict__ h,
                            const int* __restrict__ gid,
                            float* __restrict__ hg) {
    int tid = blockIdx.x * blockDim.x + threadIdx.x;
    int node = tid >> 5;
    if (node >= N_NODES) return;
    int c = (tid & 31) * 4;
    int g = gid[node];
    float4 v = *(const float4*)(h + (size_t)node * HDIM + c);
    float* p = hg + (size_t)g * HDIM + c;
    unsafeAtomicAdd(p + 0, v.x);
    unsafeAtomicAdd(p + 1, v.y);
    unsafeAtomicAdd(p + 2, v.z);
    unsafeAtomicAdd(p + 3, v.w);
}

__global__ void cls1_kernel(const float* __restrict__ hg,
                            const float* __restrict__ wc1,
                            const float* __restrict__ bc1,
                            float* __restrict__ t1) {
    int tid = blockIdx.x * blockDim.x + threadIdx.x;   // 128*128
    int g = tid >> 7, j = tid & 127;
    float s = bc1[j];
    #pragma unroll 4
    for (int k = 0; k < HDIM; ++k) s += hg[g * HDIM + k] * wc1[k * HDIM + j];
    t1[tid] = s > 0.0f ? s : 0.0f;
}

__global__ void cls2_kernel(const float* __restrict__ t1,
                            const float* __restrict__ wc2,
                            const float* __restrict__ bc2,
                            float* __restrict__ out) {
    int tid = blockIdx.x * blockDim.x + threadIdx.x;
    if (tid >= GDIM * CDIM) return;
    int g = tid / CDIM, c = tid % CDIM;
    float s = bc2[c];
    #pragma unroll 4
    for (int k = 0; k < HDIM; ++k) s += t1[g * HDIM + k] * wc2[k * CDIM + c];
    out[tid] = s;
}

extern "C" void kernel_launch(void* const* d_in, const int* in_sizes, int n_in,
                              void* d_out, int out_size, void* d_ws, size_t ws_size,
                              hipStream_t stream) {
    const float* features = (const float*)d_in[0];
    const int*   src      = (const int*)d_in[1];
    const int*   dst      = (const int*)d_in[2];
    const int*   gid      = (const int*)d_in[3];
    const float* eps      = (const float*)d_in[4];
    const float* w1[3] = { (const float*)d_in[5],  (const float*)d_in[9],  (const float*)d_in[13] };
    const float* b1[3] = { (const float*)d_in[6],  (const float*)d_in[10], (const float*)d_in[14] };
    const float* w2[3] = { (const float*)d_in[7],  (const float*)d_in[11], (const float*)d_in[15] };
    const float* b2[3] = { (const float*)d_in[8],  (const float*)d_in[12], (const float*)d_in[16] };
    const float* wc1 = (const float*)d_in[17];
    const float* bc1 = (const float*)d_in[18];
    const float* wc2 = (const float*)d_in[19];
    const float* bc2 = (const float*)d_in[20];
    float* out = (float*)d_out;

    const size_t NH = (size_t)N_NODES * HDIM;
    char* ws = (char*)d_ws;
    float* agg  = (float*)ws; ws += NH * sizeof(float);
    float* bufA = (float*)ws; ws += NH * sizeof(float);
    float* bufB = (float*)ws; ws += NH * sizeof(float);
    float* hg   = (float*)ws; ws += (size_t)GDIM * HDIM * sizeof(float);
    float* t1   = (float*)ws; ws += (size_t)GDIM * HDIM * sizeof(float);
    unsigned int* wpk[6];
    for (int i = 0; i < 6; ++i) { wpk[i] = (unsigned int*)ws; ws += 64 * HDIM * sizeof(unsigned int); }

    // Pack all MLP weights to bf16 B-fragment layout (reused ~1563x each)
    for (int i = 0; i < 3; ++i) {
        pack_weight_kernel<<<32, 256, 0, stream>>>(w1[i], wpk[2 * i]);
        pack_weight_kernel<<<32, 256, 0, stream>>>(w2[i], wpk[2 * i + 1]);
    }

    const float* hin = features;
    float* houts[3] = { bufA, bufB, bufA };
    const int mlp_grid = (N_NODES + 31) / 32;   // 1563
    for (int L = 0; L < 3; ++L) {
        init_agg_kernel<<<6250, 256, 0, stream>>>(hin, eps, L, agg);
        scatter_kernel<<<(N_EDGES * 32) / 256, 256, 0, stream>>>(hin, src, dst, agg);
        gin_mlp_kernel<<<mlp_grid, 256, 0, stream>>>(
            agg, wpk[2 * L], b1[L], wpk[2 * L + 1], b2[L], houts[L]);
        hin = houts[L];
    }

    zero_kernel<<<(GDIM * HDIM + 255) / 256, 256, 0, stream>>>(hg, GDIM * HDIM);
    pool_kernel<<<(N_NODES * 32) / 256, 256, 0, stream>>>(hin, gid, hg);
    cls1_kernel<<<(GDIM * HDIM) / 256, 256, 0, stream>>>(hg, wc1, bc1, t1);
    cls2_kernel<<<5, 256, 0, stream>>>(t1, wc2, bc2, out);
}
